// AttentionBlock_30408368455898
// MI455X (gfx1250) — matrix operand order repaired and checked by reading, not compiled, not run on hardware
//
#include <hip/hip_runtime.h>
#include <hip/hip_bf16.h>

// ---------- types ----------
typedef __bf16          v16bf  __attribute__((ext_vector_type(16)));
typedef float           v8f    __attribute__((ext_vector_type(8)));
typedef unsigned short  ushort8  __attribute__((ext_vector_type(8)));
typedef unsigned short  ushort16 __attribute__((ext_vector_type(16)));
typedef int             int4v    __attribute__((ext_vector_type(4)));

// fp32 -> bf16, round-to-nearest-even
__device__ __forceinline__ unsigned short f2bf(float f) {
  unsigned int u = __float_as_uint(f);
  unsigned int r = (u + 0x7FFFu + ((u >> 16) & 1u)) >> 16;
  return (unsigned short)r;
}

__device__ __forceinline__ v8f wmma_bf16(v16bf a, v16bf b, v8f c) {
  return __builtin_amdgcn_wmma_f32_16x16x32_bf16(false, a, false, b, (short)0, c, false, false);
}

// A fragment (16x32, bf16) from row-major [M][ldK] memory (global or LDS).
// lanes 0-15: row=r0+lane, halves 0-7 = K cc..cc+7, halves 8-15 = K cc+16..cc+23
// lanes16-31: row=r0+lane-16, K ranges +8.   Two 16B loads per lane.
__device__ __forceinline__ v16bf load_a_frag(const unsigned short* base, int r0,
                                             int ldK, int cc, int lane) {
  int row = r0 + (lane & 15);
  const unsigned short* p = base + (size_t)row * ldK + cc + ((lane >> 4) << 3);
  ushort8 lo = *(const ushort8*)p;
  ushort8 hi = *(const ushort8*)(p + 16);
  ushort16 t = __builtin_shufflevector(lo, hi, 0,1,2,3,4,5,6,7,8,9,10,11,12,13,14,15);
  return __builtin_bit_cast(v16bf, t);
}

// B fragment (32x16, bf16) from [N][ldK] memory (B^T row-major, K contiguous).
// lanes 0-15: col=c0+lane, halves 0-7 = K cc..cc+7, halves 8-15 = K cc+16..cc+23
// lanes16-31: K ranges +8.   Two 16B loads per lane.
__device__ __forceinline__ v16bf load_b_frag(const unsigned short* base, int c0,
                                             int ldK, int cc, int lane) {
  int col = c0 + (lane & 15);
  const unsigned short* p = base + (size_t)col * ldK + cc + ((lane >> 4) << 3);
  ushort8 lo = *(const ushort8*)p;
  ushort8 hi = *(const ushort8*)(p + 16);
  ushort16 t = __builtin_shufflevector(lo, hi, 0,1,2,3,4,5,6,7,8,9,10,11,12,13,14,15);
  return __builtin_bit_cast(v16bf, t);
}

// ---------- problem constants ----------
#define B_ 32
#define C_ 256
#define N_ 1024

// ---------- async copy to LDS (CDNA5 ASYNCcnt path), with safe fallback ----------
#if defined(__has_builtin)
#if __has_builtin(__builtin_amdgcn_global_load_async_to_lds_b128) && \
    __has_builtin(__builtin_amdgcn_s_wait_asynccnt)
#define USE_ASYNC_LDS 1
#endif
#endif

__device__ __forceinline__ void cp16(const unsigned short* g, unsigned short* l) {
#ifdef USE_ASYNC_LDS
  __builtin_amdgcn_global_load_async_to_lds_b128(
      (__attribute__((address_space(1))) int4v*)g,
      (__attribute__((address_space(3))) int4v*)l, 0, 0);
#else
  *(ushort8*)l = *(const ushort8*)g;
#endif
}

// ---------- 0) fp32 weights -> bf16 ----------
__global__ void __launch_bounds__(256)
cvt_kernel(const float* __restrict__ qkvw, const float* __restrict__ projw,
           unsigned short* __restrict__ wq, unsigned short* __restrict__ wp) {
  int i = blockIdx.x * 256 + threadIdx.x;
  if (i < 768 * 256) wq[i] = f2bf(qkvw[i]);
  if (i < 256 * 256) wp[i] = f2bf(projw[i]);
}

// ---------- 1) GroupNorm, writes h^T [b][n][c] in bf16 ----------
__global__ void __launch_bounds__(256)
gn_kernel(const float* __restrict__ x, const float* __restrict__ gamma,
          const float* __restrict__ beta, unsigned short* __restrict__ hT) {
  int b = blockIdx.x >> 5, g = blockIdx.x & 31;
  const float* xb = x + ((size_t)b * C_ + g * 8) * N_;   // 8 channels x 1024
  float s = 0.f, ss = 0.f;
  for (int i = threadIdx.x; i < 8192; i += 256) { float v = xb[i]; s += v; ss += v * v; }
#pragma unroll
  for (int off = 16; off; off >>= 1) { s += __shfl_xor(s, off, 32); ss += __shfl_xor(ss, off, 32); }
  __shared__ float red[16];
  int wid = threadIdx.x >> 5, lane = threadIdx.x & 31;
  if (lane == 0) { red[wid] = s; red[8 + wid] = ss; }
  __syncthreads();
  if (threadIdx.x == 0) {
    float ts = 0.f, tss = 0.f;
    for (int w = 0; w < 8; ++w) { ts += red[w]; tss += red[8 + w]; }
    float mean = ts * (1.f / 8192.f);
    float var  = tss * (1.f / 8192.f) - mean * mean;
    red[0] = mean; red[1] = rsqrtf(var + 1e-5f);
  }
  __syncthreads();
  float mean = red[0], rstd = red[1];
  unsigned short* hb = hT + (size_t)b * N_ * C_;
  for (int i = threadIdx.x; i < 8192; i += 256) {
    int cl = i >> 10, n = i & 1023, c = g * 8 + cl;
    float v = (xb[i] - mean) * rstd * gamma[c] + beta[c];
    hb[(size_t)n * C_ + c] = f2bf(v);
  }
}

// ---------- 2) QKV GEMM: (768x256) x (256x1024) per batch ----------
// q,k written n-major -> qkT[b][n][0..511]; v written o-major -> vbuf[b][c][m]
__global__ void __launch_bounds__(256)
qkv_kernel(const unsigned short* __restrict__ W, const float* __restrict__ bias,
           const unsigned short* __restrict__ hT,
           unsigned short* __restrict__ qkT, unsigned short* __restrict__ vbuf) {
  int wid = threadIdx.x >> 5, lane = threadIdx.x & 31;
  int tile = blockIdx.x * 8 + wid;              // 32 * 48 * 64 tiles
  int b = tile / (48 * 64); int r2 = tile % (48 * 64);
  int o0 = (r2 >> 6) * 16, n0 = (r2 & 63) * 16;
  const unsigned short* hb = hT + (size_t)b * N_ * C_;
  int half = lane >> 4, l15 = lane & 15;

  v8f acc = {};
#pragma unroll
  for (int kk = 0; kk < 8; ++kk) {
    v16bf a  = load_a_frag(W,  o0, C_, kk * 32, lane);   // W[o][c]
    v16bf bb = load_b_frag(hb, n0, C_, kk * 32, lane);   // h^T[n][c]
    acc = wmma_bf16(a, bb, acc);
  }
#pragma unroll
  for (int r = 0; r < 8; ++r) acc[r] += bias[o0 + r + half * 8];

  if (o0 < 512) {                       // q,k -> [n][o], one 16B store/lane
    ushort8 st;
#pragma unroll
    for (int r = 0; r < 8; ++r) st[r] = f2bf(acc[r]);
    int n = n0 + l15;
    *(ushort8*)(qkT + ((size_t)b * N_ + n) * 512 + o0 + half * 8) = st;
  } else {                              // v -> [c][m]
    int m = n0 + l15;
    unsigned short* vb = vbuf + (size_t)b * C_ * N_;
#pragma unroll
    for (int r = 0; r < 8; ++r) {
      int c = (o0 - 512) + r + half * 8;
      vb[(size_t)c * N_ + m] = f2bf(acc[r]);
    }
  }
}

// ---------- 3) Fused flash attention, LDS-shared K/V tiles ----------
// 8 waves/block, each owns one 16-row n-tile; block cooperatively double-buffers
// the 32xC K-tile and Cx32 V-tile in LDS (async DMA overlapped with WMMA).
__device__ __forceinline__ void stage_tiles(const unsigned short* kbase,
                                            const unsigned short* vb, int m0,
                                            unsigned short* lK, unsigned short* lV,
                                            int tid) {
#pragma unroll
  for (int k = 0; k < 4; ++k) {
    int idx = tid + k * 256;                       // 1024 chunks of 16B each
    int rK = idx >> 5, cK = (idx & 31) * 8;        // K tile: 32 x 256 halves
    cp16(kbase + (size_t)(m0 + rK) * 512 + cK, lK + rK * 256 + cK);
    int rV = idx >> 2, cV = (idx & 3) * 8;         // V tile: 256 x 32 halves
    cp16(vb + (size_t)rV * N_ + m0 + cV, lV + rV * 32 + cV);
  }
}

__global__ void __launch_bounds__(256)
attn_kernel(const unsigned short* __restrict__ qkT,
            const unsigned short* __restrict__ vbuf,
            unsigned short* __restrict__ outT) {
  __shared__ unsigned short ldsK[2][32 * 256];     // 2 x 16KB
  __shared__ unsigned short ldsV[2][256 * 32];     // 2 x 16KB
  int tid = threadIdx.x;
  int wid = tid >> 5, lane = tid & 31;
  int b = blockIdx.x >> 3;
  int n0 = ((blockIdx.x & 7) * 8 + wid) * 16;
  const unsigned short* qk = qkT + (size_t)b * N_ * 512;
  const unsigned short* kb = qk + 256;             // k columns 256..511
  const unsigned short* vb = vbuf + (size_t)b * C_ * N_;
  int half = lane >> 4, l15 = lane & 15;
  const float scale = 0.0625f;                     // 1/sqrt(256)

  v16bf qf[8];
#pragma unroll
  for (int kk = 0; kk < 8; ++kk) qf[kk] = load_b_frag(qk, n0, 512, kk * 32, lane);

  v8f acc[16];
#pragma unroll
  for (int i = 0; i < 16; ++i) acc[i] = (v8f){};

  float runmax = -1e30f, runsum = 0.f;

  stage_tiles(kb, vb, 0, ldsK[0], ldsV[0], tid);

  for (int it = 0; it < 32; ++it) {
    int cur = it & 1;
#ifdef USE_ASYNC_LDS
    __builtin_amdgcn_s_wait_asynccnt(0);
#endif
    __syncthreads();                               // publish tile[cur]
    if (it + 1 < 32)                               // prefetch under compute
      stage_tiles(kb, vb, (it + 1) * 32, ldsK[cur ^ 1], ldsV[cur ^ 1], tid);

    const unsigned short* tK = ldsK[cur];
    const unsigned short* tV = ldsV[cur];

    // sT = K^T Q : two 16x16 tiles (local rows 0..15, 16..31)
    v8f s0 = {}, s1 = {};
#pragma unroll
    for (int kk = 0; kk < 8; ++kk) {
      v16bf k0 = load_a_frag(tK, 0,  C_, kk * 32, lane);
      v16bf k1 = load_a_frag(tK, 16, C_, kk * 32, lane);
      s0 = wmma_bf16(k0, qf[kk], s0);
      s1 = wmma_bf16(k1, qf[kk], s1);
    }
    // online softmax: each lane holds 16 m-values of attention row n=l15
    float cmax = -1e30f;
#pragma unroll
    for (int r = 0; r < 8; ++r) {
      s0[r] *= scale; s1[r] *= scale;
      cmax = fmaxf(cmax, fmaxf(s0[r], s1[r]));
    }
    cmax = fmaxf(cmax, __shfl_xor(cmax, 16, 32));
    float newmax = fmaxf(runmax, cmax);
    float corr = __expf(runmax - newmax);
    runmax = newmax;

    float csum = 0.f;
    ushort16 pu;
#pragma unroll
    for (int r = 0; r < 8; ++r) {
      float p0 = __expf(s0[r] - newmax);
      float p1 = __expf(s1[r] - newmax);
      csum += p0 + p1;
      pu[r] = f2bf(p0); pu[8 + r] = f2bf(p1);      // exactly PV A-fragment order
    }
    csum += __shfl_xor(csum, 16, 32);
    runsum = runsum * corr + csum;
    v16bf pf = __builtin_bit_cast(v16bf, pu);

    // broadcast per-row correction into accumulator layout (lane=col c, vgpr=row n)
    float cb[8];
#pragma unroll
    for (int r = 0; r < 8; ++r) cb[r] = __shfl(corr, r + half * 8, 32);

#pragma unroll
    for (int ct = 0; ct < 16; ++ct) {
#pragma unroll
      for (int r = 0; r < 8; ++r) acc[ct][r] *= cb[r];
      v16bf vf = load_b_frag(tV, ct * 16, 32, 0, lane);  // V tile [c][m-local]
      acc[ct] = wmma_bf16(pf, vf, acc[ct]);
    }
  }

  float invL = 1.f / runsum;
  float ib[8];
#pragma unroll
  for (int r = 0; r < 8; ++r) ib[r] = __shfl(invL, r + half * 8, 32);

  unsigned short* ob = outT + (size_t)b * N_ * C_;
#pragma unroll
  for (int ct = 0; ct < 16; ++ct) {
#pragma unroll
    for (int r = 0; r < 8; ++r) {
      int n = n0 + r + half * 8;
      ob[(size_t)n * C_ + ct * 16 + l15] = f2bf(acc[ct][r] * ib[r]);
    }
  }
}

// ---------- 4) proj GEMM (yT = outT x projW^T) + bias + residual ----------
__global__ void __launch_bounds__(256)
proj_kernel(const unsigned short* __restrict__ PW, const float* __restrict__ pbias,
            const unsigned short* __restrict__ outT, const float* __restrict__ x,
            float* __restrict__ y) {
  int wid = threadIdx.x >> 5, lane = threadIdx.x & 31;
  int tile = blockIdx.x * 8 + wid;              // 32 * 64 * 16 tiles
  int b = tile >> 10; int r2 = tile & 1023;
  int n0 = (r2 >> 4) * 16, o0 = (r2 & 15) * 16;
  int half = lane >> 4, l15 = lane & 15;
  const unsigned short* ob = outT + (size_t)b * N_ * C_;

  v8f acc = {};
#pragma unroll
  for (int kk = 0; kk < 8; ++kk) {
    v16bf a  = load_a_frag(ob, n0, C_, kk * 32, lane);   // out^T[n][c]
    v16bf bb = load_b_frag(PW, o0, C_, kk * 32, lane);   // W[o][c] = B^T
    acc = wmma_bf16(a, bb, acc);
  }
  float pb = pbias[o0 + l15];
  int o = o0 + l15;
  const float* xb = x + ((size_t)b * C_ + o) * N_ + n0 + half * 8;
  float*       yb = y + ((size_t)b * C_ + o) * N_ + n0 + half * 8;
#pragma unroll
  for (int r = 0; r < 8; ++r) yb[r] = xb[r] + acc[r] + pb;   // 32B coalesced
}

// ---------- launch ----------
extern "C" void kernel_launch(void* const* d_in, const int* in_sizes, int n_in,
                              void* d_out, int out_size, void* d_ws, size_t ws_size,
                              hipStream_t stream) {
  const float* x     = (const float*)d_in[0];
  const float* gng   = (const float*)d_in[1];
  const float* gnb   = (const float*)d_in[2];
  const float* qkvw  = (const float*)d_in[3];
  const float* qkvb  = (const float*)d_in[4];
  const float* projw = (const float*)d_in[5];
  const float* projb = (const float*)d_in[6];
  float* out = (float*)d_out;

  char* ws = (char*)d_ws;
  unsigned short* wq  = (unsigned short*)(ws);                 // 768*256*2
  unsigned short* wp  = (unsigned short*)(ws + 393216);        // 256*256*2
  unsigned short* hT  = (unsigned short*)(ws + 524288);        // 32*1024*256*2
  unsigned short* qkT = (unsigned short*)(ws + 17301504);      // 32*1024*512*2
  unsigned short* vb  = (unsigned short*)(ws + 50855936);      // 32*256*1024*2
  unsigned short* oT  = (unsigned short*)(ws + 67633152);      // 32*1024*256*2

  cvt_kernel <<<768,  256, 0, stream>>>(qkvw, projw, wq, wp);
  gn_kernel  <<<1024, 256, 0, stream>>>(x, gng, gnb, hT);
  qkv_kernel <<<12288,256, 0, stream>>>(wq, qkvb, hT, qkT, vb);
  attn_kernel<<<256,  256, 0, stream>>>(qkT, vb, oT);
  proj_kernel<<<4096, 256, 0, stream>>>(wp, projb, oT, x, out);
}
